// MARN_34815004901646
// MI455X (gfx1250) — compile-verified, matrix operand order checked by reading, not run on hardware
//
#include <hip/hip_runtime.h>
#include <hip/hip_bf16.h>
#include <math.h>

typedef __bf16 bf16;
typedef __attribute__((ext_vector_type(16))) __bf16 v16bf;
typedef __attribute__((ext_vector_type(8)))  float  v8f;

#define TILE_M 64
#define TILE_N 64
#define KT 32
#define LDS_STRIDE 48   // 96 bytes/row: v16bf fragments stay 32B-aligned, staggered banks

// ---- CDNA5 async global->LDS copy (ASYNCcnt-tracked), 16 bytes per lane ----
__device__ __forceinline__ void async_cp16(void* lds_dst, const void* gsrc) {
  const unsigned lds = (unsigned)(unsigned long long)(uintptr_t)lds_dst; // low 32b = LDS offset
  const unsigned long long ga = (unsigned long long)(uintptr_t)gsrc;
  asm volatile("global_load_async_to_lds_b128 %0, %1, off"
               :: "v"(lds), "v"(ga)
               : "memory");
}
__device__ __forceinline__ void async_wait0() {
#if __has_builtin(__builtin_amdgcn_s_wait_asynccnt)
  __builtin_amdgcn_s_wait_asynccnt(0);
#else
  asm volatile("s_wait_asynccnt 0x0" ::: "memory");
#endif
}

// ---------------------------------------------------------------------------
// Fused GEMM: C[M][N] = sum_{p<NP} A_p @ B_p + bias, optional ReLU.
// A_p: fp32 if AF32 (NP==1 only) else bf16, row-major, leading dim lda.
// B_p: bf16 PRE-TRANSPOSED [N][K], leading dim ldb (lane K-fragments contiguous,
//      one k order shared by A and B: lanes 0-15 K 0-15, lanes 16-31 K 16-31).
// KPART: compile-time K of every part (equal across parts at all call sites).
// OUTF32: store fp32 to Cf, else bf16 to Cb. RELU: fold max(0,.) into epilogue.
// Double-buffered LDS; async-to-LDS staging overlaps v_wmma on the other buffer.
// Requires M%64==0, N%64==0, KPART%32==0 (true at every call site).
// ---------------------------------------------------------------------------
template <int NP, bool AF32, int KPART, bool OUTF32, bool RELU>
__global__ __launch_bounds__(256) void wmma_gemm_kernel(
    const void* A0, int lda0, const bf16* B0, int ldb0,
    const void* A1, int lda1, const bf16* B1, int ldb1,
    const void* A2, int lda2, const bf16* B2, int ldb2,
    const float* __restrict__ bias, float* __restrict__ Cf,
    bf16* __restrict__ Cb, int ldc)
{
  constexpr int CPP   = KPART / KT;   // chunks per part
  constexpr int TOTAL = NP * CPP;     // total K-chunks

  __shared__ alignas(32) bf16 As[2][TILE_M * LDS_STRIDE];
  __shared__ alignas(32) bf16 Bs[2][TILE_N * LDS_STRIDE];

  const int tid   = threadIdx.x;
  const int tileN = blockIdx.x * TILE_N;
  const int tileM = blockIdx.y * TILE_M;

  const int ldRow = tid >> 2;        // 0..63
  const int ldCol = (tid & 3) * 8;   // 0,8,16,24

  const int wave = tid >> 5;         // 0..7
  const int lane = tid & 31;
  const int subM = (wave >> 2) * 32; // 0 or 32
  const int subN = (wave & 3) * 16;  // 0,16,32,48
  const int kh   = (lane >> 4) * 16; // K-half held by this lane
  const int ln   = lane & 15;

  const void* Aarr[3] = {A0, A1, A2};
  const bf16* Barr[3] = {B0, B1, B2};
  const int ldaArr[3] = {lda0, lda1, lda2};
  const int ldbArr[3] = {ldb0, ldb1, ldb2};

  auto stage = [&](int p, int k0, int buf) {
    if (AF32) {
      // fp32 source: convert through VGPRs (precompute GEMM only)
      const float* ap = (const float*)Aarr[p] +
                        (size_t)(tileM + ldRow) * ldaArr[p] + k0 + ldCol;
      bf16* dst = &As[buf][ldRow * LDS_STRIDE + ldCol];
      #pragma unroll
      for (int i = 0; i < 8; ++i) dst[i] = (bf16)ap[i];
    } else {
      const bf16* ap = (const bf16*)Aarr[p] +
                       (size_t)(tileM + ldRow) * ldaArr[p] + k0 + ldCol;
      async_cp16(&As[buf][ldRow * LDS_STRIDE + ldCol], ap);
    }
    const bf16* bp = Barr[p] + (size_t)(tileN + ldRow) * ldbArr[p] + k0 + ldCol;
    async_cp16(&Bs[buf][ldRow * LDS_STRIDE + ldCol], bp);
    __builtin_prefetch(bp + 2 * KT, 0, 3);  // warm GL2 two chunks ahead
  };

  v8f acc0 = {};
  v8f acc1 = {};

  stage(0, 0, 0);
  async_wait0();
  __syncthreads();

  for (int ch = 0; ch < TOTAL; ++ch) {
    const int buf = ch & 1;
    if (ch + 1 < TOTAL) {           // stage next chunk into the other buffer
      const int nc = ch + 1;
      const int p  = nc / CPP;      // constexpr divisor -> cheap
      const int k0 = (nc - p * CPP) * KT;
      stage(p, k0, nc & 1);
    }

    v16bf a0v = *(const v16bf*)&As[buf][(subM + ln) * LDS_STRIDE + kh];
    v16bf a1v = *(const v16bf*)&As[buf][(subM + 16 + ln) * LDS_STRIDE + kh];
    v16bf bv  = *(const v16bf*)&Bs[buf][(subN + ln) * LDS_STRIDE + kh];
    acc0 = __builtin_amdgcn_wmma_f32_16x16x32_bf16(false, a0v, false, bv,
                                                   (short)0, acc0, false, false);
    acc1 = __builtin_amdgcn_wmma_f32_16x16x32_bf16(false, a1v, false, bv,
                                                   (short)0, acc1, false, false);

    if (ch + 1 < TOTAL) {
      async_wait0();      // this wave's async LDS writes landed
      __syncthreads();    // everyone done reading old buffer / writes visible
    }
  }

  // C/D layout: VGPR r, lanes 0-15 -> M=r, N=lane; lanes 16-31 -> M=8+r, N=lane-16
  const int col = tileN + subN + ln;
  const float bvv = bias[col];
  #pragma unroll
  for (int s = 0; s < 2; ++s) {
    v8f acc = s ? acc1 : acc0;
    const int baseM = tileM + subM + s * 16 + (lane >> 4) * 8;
    #pragma unroll
    for (int r = 0; r < 8; ++r) {
      float v = acc[r] + bvv;
      if constexpr (RELU) v = fmaxf(v, 0.f);
      const size_t idx = (size_t)(baseM + r) * ldc + col;
      if constexpr (OUTF32) Cf[idx] = v;
      else                  Cb[idx] = (bf16)v;
    }
  }
}

// fp32 [K][N] -> bf16 transposed [N][K]
__global__ void tconv_kernel(const float* __restrict__ src, bf16* __restrict__ dst,
                             int K, int N) {
  const int i = blockIdx.x * blockDim.x + threadIdx.x;
  if (i >= K * N) return;
  const int k = i / N;
  const int n = i % N;
  dst[(size_t)n * K + k] = (bf16)src[i];
}

__global__ void zero_kernel(unsigned int* p, int n) {
  const int i = blockIdx.x * blockDim.x + threadIdx.x;
  if (i < n) p[i] = 0;
}

// LSTHM gate math. S = [2][B][1024] fp32 (gates packed [f,i,o,ch]),
// c = [2][B][256] fp32 state, h = [2][B][256] bf16 output. grid.z selects cell.
__global__ __launch_bounds__(256) void gates_kernel(const float* __restrict__ S,
                                                    float* __restrict__ c,
                                                    bf16* __restrict__ h) {
  const int z = blockIdx.z;
  const int idx = blockIdx.x * 256 + threadIdx.x;  // 0..65535
  const int b = idx >> 8;
  const int d = idx & 255;
  const float* s = S + (size_t)z * 256 * 1024 + (size_t)b * 1024;
  const float f  = 1.f / (1.f + __expf(-s[d]));
  const float ig = 1.f / (1.f + __expf(-s[256 + d]));
  const float o  = 1.f / (1.f + __expf(-s[512 + d]));
  const float ch = tanhf(s[768 + d]);
  float* cp = c + (size_t)z * 65536 + idx;
  const float cn = f * (*cp) + ig * ch;
  *cp = cn;
  h[(size_t)z * 65536 + idx] = (bf16)(tanhf(cn) * o);
}

// Row softmax over 4096 logits, then scatter attended copies:
// v[b][j*256+d] = att[b][j*512+d]     * h_v[b][d]
// t[b][j*256+d] = att[b][j*512+256+d] * h_t[b][d]
__global__ __launch_bounds__(256) void softmax_attend_kernel(
    const float* __restrict__ attL, const bf16* __restrict__ h,
    bf16* __restrict__ vb, bf16* __restrict__ tb) {
  const int b = blockIdx.x;
  const int tid = threadIdx.x;
  __shared__ float red[256];
  float vals[16];
  const float* row = attL + (size_t)b * 4096 + tid * 16;
  float mx = -1e30f;
  #pragma unroll
  for (int i = 0; i < 16; ++i) { vals[i] = row[i]; mx = fmaxf(mx, vals[i]); }
  red[tid] = mx; __syncthreads();
  for (int s = 128; s > 0; s >>= 1) {
    if (tid < s) red[tid] = fmaxf(red[tid], red[tid + s]);
    __syncthreads();
  }
  mx = red[0]; __syncthreads();
  float sum = 0.f;
  #pragma unroll
  for (int i = 0; i < 16; ++i) { vals[i] = __expf(vals[i] - mx); sum += vals[i]; }
  red[tid] = sum; __syncthreads();
  for (int s = 128; s > 0; s >>= 1) {
    if (tid < s) red[tid] += red[tid + s];
    __syncthreads();
  }
  const float inv = 1.f / red[0];
  #pragma unroll
  for (int i = 0; i < 16; ++i) {
    const int idx = tid * 16 + i;
    const int j = idx >> 9;
    const int w = idx & 511;
    const float a = vals[i] * inv;
    if (w < 256)
      vb[(size_t)b * 1024 + j * 256 + w] =
          (bf16)(a * (float)h[(size_t)b * 256 + w]);
    else
      tb[(size_t)b * 1024 + j * 256 + (w - 256)] =
          (bf16)(a * (float)h[65536 + (size_t)b * 256 + (w - 256)]);
  }
}

// Final tiny layer: out[256][2] = x0[256][256] @ late1_w[256][2] + late1_b
__global__ void late1_kernel(const bf16* __restrict__ x0, const float* __restrict__ w,
                             const float* __restrict__ b, float* __restrict__ out) {
  const int i = blockIdx.x * blockDim.x + threadIdx.x;
  if (i >= 512) return;
  const int bb = i >> 1;
  const int n  = i & 1;
  float acc = b[n];
  for (int d = 0; d < 256; ++d) acc += (float)x0[bb * 256 + d] * w[d * 2 + n];
  out[i] = acc;
}

extern "C" void kernel_launch(void* const* d_in, const int* in_sizes, int n_in,
                              void* d_out, int out_size, void* d_ws, size_t ws_size,
                              hipStream_t stream) {
  (void)in_sizes; (void)n_in; (void)out_size; (void)ws_size;
  const float* x_visual  = (const float*)d_in[0];
  const float* x_tactile = (const float*)d_in[1];
  const float* fc0_w = (const float*)d_in[2];
  const float* fc0_b = (const float*)d_in[3];
  const float* fc1_w = (const float*)d_in[4];
  const float* fc1_b = (const float*)d_in[5];
  const float* lv_W = (const float*)d_in[6];
  const float* lv_U = (const float*)d_in[7];
  const float* lv_V = (const float*)d_in[8];
  const float* lv_b = (const float*)d_in[9];
  const float* lt_W = (const float*)d_in[10];
  const float* lt_U = (const float*)d_in[11];
  const float* lt_V = (const float*)d_in[12];
  const float* lt_b = (const float*)d_in[13];
  const float* att_w = (const float*)d_in[14];
  const float* att_b = (const float*)d_in[15];
  const float* red_v_w = (const float*)d_in[16];
  const float* red_v_b = (const float*)d_in[17];
  const float* red_t_w = (const float*)d_in[18];
  const float* red_t_b = (const float*)d_in[19];
  const float* mab_out_w = (const float*)d_in[20];
  const float* mab_out_b = (const float*)d_in[21];
  const float* late0_w = (const float*)d_in[22];
  const float* late0_b = (const float*)d_in[23];
  const float* late1_w = (const float*)d_in[24];
  const float* late1_b = (const float*)d_in[25];
  float* out = (float*)d_out;

  constexpr int B = 256, T = 25, PRE = 2048, D = 256, K4 = 4;

  char* ws = (char*)d_ws;
  size_t off = 0;
  auto alloc = [&](size_t bytes) -> void* {
    void* p = ws + off;
    off += (bytes + 255) & ~(size_t)255;
    return p;
  };

  // bf16 transposed weights [N][K]
  bf16* wT_fc0  = (bf16*)alloc((size_t)PRE * D * 2);
  bf16* wT_fc1  = (bf16*)alloc((size_t)PRE * D * 2);
  bf16* wT_lvW  = (bf16*)alloc((size_t)D * 4 * D * 2);
  bf16* wT_lvU  = (bf16*)alloc((size_t)D * 4 * D * 2);
  bf16* wT_lvV  = (bf16*)alloc((size_t)D * 4 * D * 2);
  bf16* wT_ltW  = (bf16*)alloc((size_t)D * 4 * D * 2);
  bf16* wT_ltU  = (bf16*)alloc((size_t)D * 4 * D * 2);
  bf16* wT_ltV  = (bf16*)alloc((size_t)D * 4 * D * 2);
  bf16* wT_att  = (bf16*)alloc((size_t)2 * D * K4 * 2 * D * 2);
  bf16* wT_redv = (bf16*)alloc((size_t)K4 * D * D * 2);
  bf16* wT_redt = (bf16*)alloc((size_t)K4 * D * D * 2);
  bf16* wT_mab  = (bf16*)alloc((size_t)2 * D * D * 2);
  bf16* wT_l0   = (bf16*)alloc((size_t)2 * D * D * 2);
  // bf16 activations
  bf16* xv_bf = (bf16*)alloc((size_t)B * T * D * 2);
  bf16* xt_bf = (bf16*)alloc((size_t)B * T * D * 2);
  // state (contiguous for one zero-init pass)
  float* cbuf = (float*)alloc((size_t)2 * B * D * 4);
  bf16*  hbuf = (bf16*)alloc((size_t)2 * B * D * 2);
  bf16*  zzb  = (bf16*)alloc((size_t)B * D * 2);
  // per-step scratch
  float* Sbuf = (float*)alloc((size_t)2 * B * 4 * D * 4);
  float* attL = (float*)alloc((size_t)B * K4 * 2 * D * 4);
  bf16*  vbuf = (bf16*)alloc((size_t)B * K4 * D * 2);
  bf16*  tbuf = (bf16*)alloc((size_t)B * K4 * D * 2);
  bf16*  rv   = (bf16*)alloc((size_t)B * D * 2);
  bf16*  rt   = (bf16*)alloc((size_t)B * D * 2);
  bf16*  x0   = (bf16*)alloc((size_t)B * D * 2);

  auto tconv = [&](const float* src, bf16* dst, int Kd, int Nd) {
    const int total = Kd * Nd;
    tconv_kernel<<<dim3((total + 255) / 256), dim3(256), 0, stream>>>(src, dst, Kd, Nd);
  };

  // ---- weight conversion (transposed bf16) ----
  tconv(fc0_w, wT_fc0, PRE, D);
  tconv(fc1_w, wT_fc1, PRE, D);
  tconv(lv_W, wT_lvW, D, 4 * D);
  tconv(lv_U, wT_lvU, D, 4 * D);
  tconv(lv_V, wT_lvV, D, 4 * D);
  tconv(lt_W, wT_ltW, D, 4 * D);
  tconv(lt_U, wT_ltU, D, 4 * D);
  tconv(lt_V, wT_ltV, D, 4 * D);
  tconv(att_w, wT_att, 2 * D, K4 * 2 * D);
  tconv(red_v_w, wT_redv, K4 * D, D);
  tconv(red_t_w, wT_redt, K4 * D, D);
  tconv(mab_out_w, wT_mab, 2 * D, D);
  tconv(late0_w, wT_l0, 2 * D, D);

  // ---- zero initial state (c, h, zz contiguous) ----
  {
    const int zwords = (2 * B * D * 4 + 2 * B * D * 2 + B * D * 2) / 4;
    zero_kernel<<<dim3((zwords + 255) / 256), dim3(256), 0, stream>>>(
        (unsigned int*)cbuf, zwords);
  }

  const dim3 blk(256);

  // ---- precompute xv = x_visual @ fc0 + b, xt = x_tactile @ fc1 + b (bf16 out) ----
  wmma_gemm_kernel<1, true, 2048, false, false>
      <<<dim3(D / 64, (B * T) / 64), blk, 0, stream>>>(
          x_visual, PRE, wT_fc0, PRE, nullptr, 0, nullptr, 0,
          nullptr, 0, nullptr, 0, fc0_b, nullptr, xv_bf, D);
  wmma_gemm_kernel<1, true, 2048, false, false>
      <<<dim3(D / 64, (B * T) / 64), blk, 0, stream>>>(
          x_tactile, PRE, wT_fc1, PRE, nullptr, 0, nullptr, 0,
          nullptr, 0, nullptr, 0, fc1_b, nullptr, xt_bf, D);

  bf16* h_v = hbuf;
  bf16* h_t = hbuf + (size_t)B * D;

  // ---- recurrent scan ----
  for (int t = 0; t < T; ++t) {
    const bf16* xv_t = xv_bf + (size_t)t * D;  // row b lives at (b*T + t)*D
    const bf16* xt_t = xt_bf + (size_t)t * D;
    // S_v = xv_t@W + h_v@U + zz@V + b   [B x 1024] (fp32 out)
    wmma_gemm_kernel<3, false, 256, true, false>
        <<<dim3(4 * D / 64, B / 64), blk, 0, stream>>>(
            xv_t, T * D, wT_lvW, D, h_v, D, wT_lvU, D,
            zzb, D, wT_lvV, D, lv_b, Sbuf, nullptr, 4 * D);
    // S_t
    wmma_gemm_kernel<3, false, 256, true, false>
        <<<dim3(4 * D / 64, B / 64), blk, 0, stream>>>(
            xt_t, T * D, wT_ltW, D, h_t, D, wT_ltU, D,
            zzb, D, wT_ltV, D, lt_b, Sbuf + (size_t)B * 4 * D, nullptr, 4 * D);
    // gates -> new c, h (both cells)
    gates_kernel<<<dim3(B * D / 256, 1, 2), blk, 0, stream>>>(Sbuf, cbuf, hbuf);
    // attention logits: [h_v|h_t] @ att_w + att_b   [B x 2048] (fp32 out)
    wmma_gemm_kernel<2, false, 256, true, false>
        <<<dim3(K4 * 2 * D / 64, B / 64), blk, 0, stream>>>(
            h_v, D, wT_att, 2 * D, h_t, D, wT_att + D, 2 * D,
            nullptr, 0, nullptr, 0, att_b, attL, nullptr, K4 * 2 * D);
    // softmax + attended copies -> v, t (bf16)
    softmax_attend_kernel<<<dim3(B), blk, 0, stream>>>(attL, hbuf, vbuf, tbuf);
    // rv = relu(v @ red_v + b), rt = relu(t @ red_t + b)   (bf16 out, relu)
    wmma_gemm_kernel<1, false, 1024, false, true>
        <<<dim3(D / 64, B / 64), blk, 0, stream>>>(
            vbuf, K4 * D, wT_redv, K4 * D, nullptr, 0, nullptr, 0,
            nullptr, 0, nullptr, 0, red_v_b, nullptr, rv, D);
    wmma_gemm_kernel<1, false, 1024, false, true>
        <<<dim3(D / 64, B / 64), blk, 0, stream>>>(
            tbuf, K4 * D, wT_redt, K4 * D, nullptr, 0, nullptr, 0,
            nullptr, 0, nullptr, 0, red_t_b, nullptr, rt, D);
    // zz = [rv|rt] @ mab_out + b   (bf16 out)
    wmma_gemm_kernel<2, false, 256, false, false>
        <<<dim3(D / 64, B / 64), blk, 0, stream>>>(
            rv, D, wT_mab, 2 * D, rt, D, wT_mab + D, 2 * D,
            nullptr, 0, nullptr, 0, mab_out_b, nullptr, zzb, D);
  }

  // ---- classifier head ----
  wmma_gemm_kernel<2, false, 256, false, false>
      <<<dim3(D / 64, B / 64), blk, 0, stream>>>(
          h_v, D, wT_l0, 2 * D, h_t, D, wT_l0 + D, 2 * D,
          nullptr, 0, nullptr, 0, late0_b, nullptr, x0, D);
  late1_kernel<<<dim3(2), blk, 0, stream>>>(x0, late1_w, late1_b, out);
}
